// attention_node_to_edge_19258633355661
// MI455X (gfx1250) — matrix operand order repaired and checked by reading, not compiled, not run on hardware
//
#include <hip/hip_runtime.h>
#include <hip/hip_fp16.h>

// MI455X / gfx1250 fused kernel for:
//   q = x@Wq^T+bq ; k = e@Wk^T+bk ; v = e@Wv^T+bv
//   att = softmax((q.k)/8) ; x_out = att@v ; return (x_out, e)
// Key optimizations:
//  * K-projection eliminated algebraically: logit[m] = e[m].g (+const), g = Wk^T q.
//    (const = q.bk is softmax-invariant -> dropped). Logits stay full f32.
//  * V-projection done with v_wmma_f32_16x16x32_f16 (f32 accumulate).
//  * e is read from HBM exactly once; the mandatory copy of e into d_out is
//    fused into the same pass with non-temporal load/store.
//  ~1.07 GB of HBM traffic total -> ~46 us roofline at 23.3 TB/s.

typedef __attribute__((ext_vector_type(16))) _Float16 v16h;
typedef __attribute__((ext_vector_type(8)))  _Float16 v8h;
typedef __attribute__((ext_vector_type(4)))  _Float16 v4h;
typedef __attribute__((ext_vector_type(8)))  float    v8f;
typedef __attribute__((ext_vector_type(4)))  float    v4f;

#define B_   4
#define N_   512
#define D_   128
#define DH_  64
#define NWAVE 8
#define MT_PER_WAVE 4   // 32 m-tiles (512/16) spread over 8 waves

__global__ __launch_bounds__(256) void attn_n2e_kernel(
    const float* __restrict__ x,  const float* __restrict__ e,
    const float* __restrict__ Wq, const float* __restrict__ bq,
    const float* __restrict__ Wk, const float* __restrict__ bk,
    const float* __restrict__ Wv, const float* __restrict__ bv,
    float* __restrict__ out_x, float* __restrict__ out_e)
{
  // ---- LDS (~116.5 KB) ----
  __shared__ __align__(16) _Float16 Es[NWAVE][16][D_];   // 32 KB per-wave e-tile (f16)
  __shared__ __align__(16) _Float16 Vs[N_][DH_];         // 64 KB V = e@Wv^T+bv (f16)
  __shared__ __align__(32) _Float16 Wv16[DH_][D_];       // 16 KB Wv in f16, [d][c]
  __shared__ float Ls[N_];                               // logits -> exp weights
  __shared__ float red[256];
  __shared__ float xs[D_];
  __shared__ float qs[DH_];
  __shared__ float gs[D_];                               // g = Wk^T q
  __shared__ float bvs[DH_];

  const int t    = threadIdx.x;
  const int lane = t & 31;
  const int wv   = t >> 5;
  const int bid  = blockIdx.x;            // bid = b*N + n

  (void)bk; // q.bk is constant over m -> softmax invariant -> dropped exactly

  // ---------------- prologue: q, g, Wv->f16, bv ----------------
  if (t < D_)  xs[t] = x[(size_t)bid * D_ + t];
  for (int i = t; i < DH_ * D_; i += 256)
    (&Wv16[0][0])[i] = (_Float16)Wv[i];
  if (t < DH_) bvs[t] = bv[t];
  __syncthreads();

  if (t < DH_) {
    float a = bq[t];
    const float* wr = Wq + t * D_;
    #pragma unroll 4
    for (int c = 0; c < D_; ++c) a += wr[c] * xs[c];
    qs[t] = a;
  }
  __syncthreads();

  if (t < D_) {
    float a = 0.f;
    #pragma unroll 4
    for (int d = 0; d < DH_; ++d) a += qs[d] * Wk[d * D_ + t];
    gs[t] = a;
  }
  __syncthreads();

  // ---------------- main pass over e (single HBM read) ----------------
  const float inv_sqrt_dh = 0.125f;       // 1/sqrt(64)
  const int lo = lane & 15, hi = lane >> 4;

  for (int tt = 0; tt < MT_PER_WAVE; ++tt) {
    const int mt = wv + tt * NWAVE;                       // m-tile 0..31
    const size_t rowbase = ((size_t)bid * N_ + (size_t)mt * 16) * D_;

    // stage 16x128 f32 tile: NT copy out, f32 logit vs g, f16 into Es[wv]
    #pragma unroll
    for (int i = 0; i < 16; ++i) {
      const v4f val = __builtin_nontemporal_load(
          (const v4f*)(e + rowbase + (size_t)i * D_) + lane);
      __builtin_nontemporal_store(val,
          (v4f*)(out_e + rowbase + (size_t)i * D_) + lane);
      float p = val[0] * gs[4 * lane + 0] + val[1] * gs[4 * lane + 1]
              + val[2] * gs[4 * lane + 2] + val[3] * gs[4 * lane + 3];
      #pragma unroll
      for (int off = 16; off > 0; off >>= 1) p += __shfl_xor(p, off, 32);
      if (lane == 0) Ls[mt * 16 + i] = p * inv_sqrt_dh;
      v4h h;
      h[0] = (_Float16)val[0]; h[1] = (_Float16)val[1];
      h[2] = (_Float16)val[2]; h[3] = (_Float16)val[3];
      *(v4h*)&Es[wv][i][lane * 4] = h;
    }

    // V tile (16m x 64d) = Es(16x128) x Wv^T(128x64) + bv  via WMMA f16
    v8f acc[4];
    #pragma unroll
    for (int dt = 0; dt < 4; ++dt) {
      const float bvv = bvs[dt * 16 + lo];    // C/D layout: N-col = lane&15
      #pragma unroll
      for (int r = 0; r < 8; ++r) acc[dt][r] = bvv;
    }
    #pragma unroll
    for (int ks = 0; ks < 4; ++ks) {
      // A frag (16x32 f16): lane -> row lane&15; K = ks*32 + {8*hi..+7, 16+8*hi..+7}
      const _Float16* arow = &Es[wv][lo][ks * 32 + 8 * hi];
      v8h alo_ = *(const v8h*)arow;
      v8h ahi_ = *(const v8h*)(arow + 16);
      v16h afrag = __builtin_shufflevector(alo_, ahi_,
          0,1,2,3,4,5,6,7,8,9,10,11,12,13,14,15);
      #pragma unroll
      for (int dt = 0; dt < 4; ++dt) {
        // B frag (32x16): lane -> col N = dt*16 + (lane&15); K = ks*32 + {8*hi..+7, 16+8*hi..+7}
        // B[K][N] = Wv^T[c][d] = Wv[d][c] -> two 16B chunks in Wv16[d][.]
        const _Float16* brow = &Wv16[dt * 16 + lo][ks * 32 + 8 * hi];
        const v16h bfrag = __builtin_shufflevector(*(const v8h*)brow, *(const v8h*)(brow + 16),
            0,1,2,3,4,5,6,7,8,9,10,11,12,13,14,15);
        acc[dt] = __builtin_amdgcn_wmma_f32_16x16x32_f16(
            false, afrag, false, bfrag, (short)0, acc[dt], false, false);
      }
    }
    // C/D layout: VGPR r, lanes 0-15 -> M=r ; lanes 16-31 -> M=r+8
    #pragma unroll
    for (int dt = 0; dt < 4; ++dt) {
      #pragma unroll
      for (int r = 0; r < 8; ++r)
        Vs[mt * 16 + 8 * hi + r][dt * 16 + lo] = (_Float16)acc[dt][r];
    }
  }
  __syncthreads();

  // ---------------- softmax over 512 logits ----------------
  const float l0 = Ls[t], l1 = Ls[t + 256];
  red[t] = fmaxf(l0, l1);
  __syncthreads();
  for (int s = 128; s > 0; s >>= 1) {
    if (t < s) red[t] = fmaxf(red[t], red[t + s]);
    __syncthreads();
  }
  const float mx = red[0];
  __syncthreads();
  const float e0 = __expf(l0 - mx), e1 = __expf(l1 - mx);
  Ls[t] = e0; Ls[t + 256] = e1;
  red[t] = e0 + e1;
  __syncthreads();
  for (int s = 128; s > 0; s >>= 1) {
    if (t < s) red[t] += red[t + s];
    __syncthreads();
  }
  const float inv = 1.0f / red[0];
  __syncthreads();

  // ---------------- x_out[d] = sum_m att[m] * V[m][d] ----------------
  const int d = t & 63, part = t >> 6;        // 4 partial sums per d
  const int m0 = part * 128;
  float s = 0.f;
  #pragma unroll 4
  for (int m = 0; m < 128; ++m) s += Ls[m0 + m] * (float)Vs[m0 + m][d];
  red[t] = s;
  __syncthreads();
  if (t < 64) {
    out_x[(size_t)bid * DH_ + t] =
        (red[t] + red[t + 64] + red[t + 128] + red[t + 192]) * inv;
  }
}

extern "C" void kernel_launch(void* const* d_in, const int* in_sizes, int n_in,
                              void* d_out, int out_size, void* d_ws, size_t ws_size,
                              hipStream_t stream) {
  (void)in_sizes; (void)n_in; (void)out_size; (void)d_ws; (void)ws_size;
  const float* x  = (const float*)d_in[0];
  const float* e  = (const float*)d_in[1];
  const float* Wq = (const float*)d_in[2];
  const float* bq = (const float*)d_in[3];
  const float* Wk = (const float*)d_in[4];
  const float* bk = (const float*)d_in[5];
  const float* Wv = (const float*)d_in[6];
  const float* bv = (const float*)d_in[7];
  float* out_x = (float*)d_out;
  float* out_e = out_x + (size_t)B_ * N_ * DH_;   // tuple: (x_out, e) concatenated

  attn_n2e_kernel<<<B_ * N_, 256, 0, stream>>>(
      x, e, Wq, bq, Wk, bk, Wv, bv, out_x, out_e);
}
